// SketchRNN_57028575756465
// MI455X (gfx1250) — matrix operand order repaired and checked by reading, not compiled, not run on hardware
//
#include <hip/hip_runtime.h>
#include <hip/hip_bf16.h>

typedef __bf16 bf16_t;
typedef __attribute__((ext_vector_type(16))) __bf16 v16bf_t;
typedef __attribute__((ext_vector_type(8)))  float  v8f_t;

namespace {
constexpr int kB = 128, kT = 512, kD = 5, kH = 512, kZ = 128;
constexpr int k4H = 2048;
constexpr int kNWG = 32;              // workgroups per direction
constexpr int kTotalWG = 2 * kNWG;    // grid size of persistent kernel
constexpr float kEps = 1e-5f;

// workspace layout (bytes)
constexpr size_t OFF_BAR  = 0;                       // unsigned barrier counter
constexpr size_t OFF_GST  = 256;                     // float [2 phase][2 dir][128 row][2] = 4 KB
constexpr size_t OFF_CST  = OFF_GST + 4096;          // float [2][2][128][2]              = 4 KB
constexpr size_t OFF_HBUF = 16384;                   // bf16  [2 dir][128][512]           = 256 KB
constexpr size_t OFF_XST  = OFF_HBUF + 262144;       // float [2 dir][512 t][128 b][2]    = 1 MB
constexpr size_t ZERO_BYTES = OFF_XST;               // zero barrier + stats + hbuf each call
}

union FragU { v16bf_t v; uint4 q[2]; };

__device__ inline float sigmoidf_(float v) { return 1.0f / (1.0f + __expf(-v)); }

// sum across the 16 lanes that share a matrix row (keeps lanes 0-15 / 16-31 separate)
__device__ inline float red16_(float v) {
    v += __shfl_xor(v, 1, 32);
    v += __shfl_xor(v, 2, 32);
    v += __shfl_xor(v, 4, 32);
    v += __shfl_xor(v, 8, 32);
    return v;
}

__device__ inline void grid_barrier(unsigned* cnt, unsigned target) {
    __threadfence();                 // release: flush this thread's global stores
    __syncthreads();
    if (threadIdx.x == 0) {
        atomicAdd(cnt, 1u);
        while (__hip_atomic_load(cnt, __ATOMIC_ACQUIRE, __HIP_MEMORY_SCOPE_AGENT) < target) {
            __builtin_amdgcn_s_sleep(2);
        }
    }
    __syncthreads();
    __threadfence();                 // acquire: invalidate L0 so fresh L2 data is seen
}

// -------------------------------------------------------------------------
// Kernel 1: per-(dir,t,b) LayerNorm statistics of xW (K=5), stored {mean, rstd}
// -------------------------------------------------------------------------
__global__ void xstats_kernel(const float* __restrict__ x,
                              const float* __restrict__ Wx_f,
                              const float* __restrict__ Wx_b,
                              float* __restrict__ xstats) {
    const int b = blockIdx.x, t = blockIdx.y, dir = blockIdx.z;
    const float* Wx = dir ? Wx_b : Wx_f;
    float xr[kD];
#pragma unroll
    for (int d = 0; d < kD; ++d)
        xr[d] = x[((size_t)b * kT + t) * kD + (dir ? (kD - 1 - d) : d)];

    float s = 0.f, ss = 0.f;
    for (int n = threadIdx.x; n < k4H; n += 256) {
        float v = 0.f;
#pragma unroll
        for (int d = 0; d < kD; ++d) v += xr[d] * Wx[(size_t)d * k4H + n];
        s += v; ss += v * v;
    }
#pragma unroll
    for (int m = 16; m >= 1; m >>= 1) { s += __shfl_xor(s, m, 32); ss += __shfl_xor(ss, m, 32); }
    __shared__ float2 red[8];
    if ((threadIdx.x & 31) == 0) red[threadIdx.x >> 5] = make_float2(s, ss);
    __syncthreads();
    if (threadIdx.x == 0) {
        float a = 0.f, aa = 0.f;
#pragma unroll
        for (int w = 0; w < 8; ++w) { a += red[w].x; aa += red[w].y; }
        float mean = a / (float)k4H;
        float var  = aa / (float)k4H - mean * mean;
        float2 out = make_float2(mean, rsqrtf(var + kEps));
        *(float2*)(xstats + (((size_t)dir * kT + t) * kB + b) * 2) = out;
    }
}

// -------------------------------------------------------------------------
// Kernel 2: persistent LayerNorm-LSTM, 64 WGs (32 per direction), 3 grid
// barriers per time step. WG owns 16 h-columns (i/f/g/o gate blocks).
// -------------------------------------------------------------------------
__global__ void __launch_bounds__(256, 1)
lnlstm_kernel(const float* __restrict__ x,
              const float* __restrict__ Wh_f, const float* __restrict__ Wh_b,
              const float* __restrict__ Wx_f, const float* __restrict__ Wx_b,
              const float* __restrict__ b_f,  const float* __restrict__ b_b,
              const float* __restrict__ gx_f, const float* __restrict__ bx_f,
              const float* __restrict__ gx_b, const float* __restrict__ bx_b,
              const float* __restrict__ gh_f, const float* __restrict__ bh_f,
              const float* __restrict__ gh_b, const float* __restrict__ bh_b,
              const float* __restrict__ gc_f, const float* __restrict__ bc_f,
              const float* __restrict__ gc_b, const float* __restrict__ bc_b,
              unsigned* __restrict__ bar,
              float* __restrict__ gstats, float* __restrict__ cstats,
              const float* __restrict__ xstats,
              bf16_t* __restrict__ hbuf) {
    extern __shared__ char smem_raw[];
    uint4* Bpack = (uint4*)smem_raw;   // [kt(16)][nt(4)][lane(32)] x 32B fragment

    const int dir   = blockIdx.x >> 5;
    const int wg    = blockIdx.x & 31;
    const int wbase = wg * 16;
    const int wave  = threadIdx.x >> 5;
    const int lane  = threadIdx.x & 31;
    const int lmod  = lane & 15;
    const int hh    = lane >> 4;

    const float* Wh  = dir ? Wh_b : Wh_f;
    const float* Wx  = dir ? Wx_b : Wx_f;
    const float* bb  = dir ? b_b  : b_f;
    const float* gxp = dir ? gx_b : gx_f;
    const float* bxp = dir ? bx_b : bx_f;
    const float* ghp = dir ? gh_b : gh_f;
    const float* bhp = dir ? bh_b : bh_f;
    const float* gcp = dir ? gc_b : gc_f;
    const float* bcp = dir ? bc_b : bc_f;

    // ---- pack Wh slice into LDS in WMMA-B fragment layout (once) ----
    for (int e = threadIdx.x; e < 16 * 4 * 32; e += 256) {
        const int l  = e & 31;
        const int nt = (e >> 5) & 3;
        const int kt = e >> 7;
        const int n  = nt * kH + wbase + (l & 15);
        const int k0 = kt * 32 + ((l < 16) ? 0 : 16);
        FragU fr;
#pragma unroll
        for (int v = 0; v < 8; ++v) {
            fr.v[2 * v]     = (bf16_t)Wh[(size_t)(k0 + 2 * v)     * k4H + n];
            fr.v[2 * v + 1] = (bf16_t)Wh[(size_t)(k0 + 2 * v + 1) * k4H + n];
        }
        Bpack[e * 2]     = fr.q[0];
        Bpack[e * 2 + 1] = fr.q[1];
    }
    __syncthreads();

    // ---- per-lane static parameters (gate columns n = nt*512 + wbase + lmod) ----
    float p_gh[4], p_bh[4], p_gx[4], p_bx[4], p_b[4], p_wx[4][kD];
#pragma unroll
    for (int nt = 0; nt < 4; ++nt) {
        const int n = nt * kH + wbase + lmod;
        p_gh[nt] = ghp[n]; p_bh[nt] = bhp[n];
        p_gx[nt] = gxp[n]; p_bx[nt] = bxp[n];
        p_b[nt]  = bb[n];
#pragma unroll
        for (int d = 0; d < kD; ++d)
            p_wx[nt][d] = Wx[(size_t)(dir ? (kD - 1 - d) : d) * k4H + n];
    }
    const float p_gc = gcp[wbase + lmod];
    const float p_bc = bcp[wbase + lmod];

    float c_st[8];
#pragma unroll
    for (int r = 0; r < 8; ++r) c_st[r] = 0.f;

    unsigned bidx = 0;
    for (int t = 0; t < kT; ++t) {
        const int ph = t & 1;

        // ================= Phase A: recurrent GEMM via WMMA =================
        v8f_t acc[4];
#pragma unroll
        for (int nt = 0; nt < 4; ++nt)
#pragma unroll
            for (int i = 0; i < 8; ++i) acc[nt][i] = 0.f;

        const bf16_t* hsrc = hbuf + (size_t)dir * kB * kH + (size_t)(wave * 16 + lmod) * kH;
#pragma unroll 4
        for (int kt = 0; kt < 16; ++kt) {
            FragU a;
            const int kofs = kt * 32 + hh * 16;    // A layout: lanes<16 K{0..15}
            a.q[0] = *(const uint4*)(hsrc + kofs);
            a.q[1] = *(const uint4*)(hsrc + kofs + 8);
            int bbi = (kt * 4 * 32 + lane) * 2;
#pragma unroll
            for (int nt = 0; nt < 4; ++nt) {
                FragU bf_;
                bf_.q[0] = Bpack[bbi];
                bf_.q[1] = Bpack[bbi + 1];
                bbi += 64;
                acc[nt] = __builtin_amdgcn_wmma_f32_16x16x32_bf16(
                    false, a.v, false, bf_.v, (short)0, acc[nt], false, false);
            }
        }

        // partial LN stats of gate pre-activations (per row, over this WG's 64 cols)
        float* gph = gstats + (size_t)(ph * 2 + dir) * kB * 2;
#pragma unroll
        for (int r = 0; r < 8; ++r) {
            float s  = acc[0][r] + acc[1][r] + acc[2][r] + acc[3][r];
            float ss = acc[0][r] * acc[0][r] + acc[1][r] * acc[1][r] +
                       acc[2][r] * acc[2][r] + acc[3][r] * acc[3][r];
            s = red16_(s); ss = red16_(ss);
            if (lmod == 0) {
                const int row = wave * 16 + hh * 8 + r;
                atomicAdd(gph + row * 2,     s);
                atomicAdd(gph + row * 2 + 1, ss);
            }
        }
        grid_barrier(bar, (++bidx) * kTotalWG);     // B1

        // ================= Phase B: LN + gates + c update =================
        float oga[8];
        float* cph = cstats + (size_t)(ph * 2 + dir) * kB * 2;
#pragma unroll
        for (int r = 0; r < 8; ++r) {
            const int row = wave * 16 + hh * 8 + r;
            float2 gv = *(const float2*)(gph + row * 2);
            const float gmean = gv.x / (float)k4H;
            const float grstd = rsqrtf(gv.y / (float)k4H - gmean * gmean + kEps);
            float2 xs = *(const float2*)(xstats + (((size_t)dir * kT + t) * kB + row) * 2);
            float xrow[kD];
#pragma unroll
            for (int d = 0; d < kD; ++d) xrow[d] = x[((size_t)row * kT + t) * kD + d];

            float pre[4];
#pragma unroll
            for (int nt = 0; nt < 4; ++nt) {
                float xw = 0.f;
#pragma unroll
                for (int d = 0; d < kD; ++d) xw += xrow[d] * p_wx[nt][d];
                const float lnx = p_gx[nt] * (xw - xs.x) * xs.y + p_bx[nt];
                const float lnh = p_gh[nt] * (acc[nt][r] - gmean) * grstd + p_bh[nt];
                pre[nt] = lnx + lnh + p_b[nt];
            }
            const float ig = sigmoidf_(pre[0]);
            const float fg = sigmoidf_(pre[1]);
            const float gg = tanhf(pre[2]);
            oga[r] = sigmoidf_(pre[3]);
            const float c = fg * c_st[r] + ig * gg;
            c_st[r] = c;
            float s = red16_(c), ss = red16_(c * c);
            if (lmod == 0) {
                atomicAdd(cph + row * 2,     s);
                atomicAdd(cph + row * 2 + 1, ss);
            }
        }
        if (wg == 0)   // zero next-phase gate stats (safe: last read 3 barriers ago)
            gstats[(size_t)(((ph ^ 1) * 2 + dir) * kB) * 2 + threadIdx.x] = 0.f;
        grid_barrier(bar, (++bidx) * kTotalWG);     // B2

        // ================= Phase C: LN(c) + h out =================
        bf16_t* hdst = hbuf + (size_t)dir * kB * kH + wbase + lmod;
#pragma unroll
        for (int r = 0; r < 8; ++r) {
            const int row = wave * 16 + hh * 8 + r;
            float2 cv = *(const float2*)(cph + row * 2);
            const float cmean = cv.x / (float)kH;
            const float crstd = rsqrtf(cv.y / (float)kH - cmean * cmean + kEps);
            const float lnc = p_gc * (c_st[r] - cmean) * crstd + p_bc;
            const float hv  = oga[r] * tanhf(lnc);
            hdst[(size_t)row * kH] = (bf16_t)hv;
        }
        if (wg == 0)
            cstats[(size_t)(((ph ^ 1) * 2 + dir) * kB) * 2 + threadIdx.x] = 0.f;
        grid_barrier(bar, (++bidx) * kTotalWG);     // B3: h published
    }
}

// -------------------------------------------------------------------------
// Kernel 3: VAE head  z = mu + exp((h@Wsig+bsig)/2) * eps
// -------------------------------------------------------------------------
__global__ void head_kernel(const bf16_t* __restrict__ hbuf,
                            const float* __restrict__ Wmu, const float* __restrict__ bmu,
                            const float* __restrict__ Wsg, const float* __restrict__ bsg,
                            const float* __restrict__ eps, float* __restrict__ out) {
    const int b = blockIdx.x, zc = threadIdx.x;
    const bf16_t* hf = hbuf + (size_t)b * kH;
    const bf16_t* hb = hbuf + (size_t)kB * kH + (size_t)b * kH;
    float mu = 0.f, sg = 0.f;
    for (int k = 0; k < kH; ++k) {
        const float hv = (float)hf[k];
        mu += hv * Wmu[(size_t)k * kZ + zc];
        sg += hv * Wsg[(size_t)k * kZ + zc];
    }
    for (int k = 0; k < kH; ++k) {
        const float hv = (float)hb[k];
        mu += hv * Wmu[(size_t)(kH + k) * kZ + zc];
        sg += hv * Wsg[(size_t)(kH + k) * kZ + zc];
    }
    mu += bmu[zc];
    const float sigma = __expf((sg + bsg[zc]) * 0.5f);
    out[(size_t)b * kZ + zc] = mu + sigma * eps[(size_t)b * kZ + zc];
}

extern "C" void kernel_launch(void* const* d_in, const int* in_sizes, int n_in,
                              void* d_out, int out_size, void* d_ws, size_t ws_size,
                              hipStream_t stream) {
    (void)in_sizes; (void)n_in; (void)out_size; (void)ws_size;
    const float* x     = (const float*)d_in[0];
    const float* eps   = (const float*)d_in[1];
    const float* Wx_f  = (const float*)d_in[2];
    const float* Wh_f  = (const float*)d_in[3];
    const float* b_f   = (const float*)d_in[4];
    const float* gx_f  = (const float*)d_in[5];
    const float* bx_f  = (const float*)d_in[6];
    const float* gh_f  = (const float*)d_in[7];
    const float* bh_f  = (const float*)d_in[8];
    const float* gc_f  = (const float*)d_in[9];
    const float* bc_f  = (const float*)d_in[10];
    const float* Wx_b  = (const float*)d_in[11];
    const float* Wh_b  = (const float*)d_in[12];
    const float* b_b   = (const float*)d_in[13];
    const float* gx_b  = (const float*)d_in[14];
    const float* bx_b  = (const float*)d_in[15];
    const float* gh_b  = (const float*)d_in[16];
    const float* bh_b  = (const float*)d_in[17];
    const float* gc_b  = (const float*)d_in[18];
    const float* bc_b  = (const float*)d_in[19];
    const float* W_mu  = (const float*)d_in[20];
    const float* b_mu  = (const float*)d_in[21];
    const float* W_sig = (const float*)d_in[22];
    const float* b_sig = (const float*)d_in[23];

    unsigned char* w = (unsigned char*)d_ws;
    unsigned* bar   = (unsigned*)(w + OFF_BAR);
    float*   gst    = (float*)(w + OFF_GST);
    float*   cst    = (float*)(w + OFF_CST);
    bf16_t*  hbuf   = (bf16_t*)(w + OFF_HBUF);
    float*   xst    = (float*)(w + OFF_XST);

    hipMemsetAsync(w, 0, ZERO_BYTES, stream);   // barrier + stats + h0 = 0

    xstats_kernel<<<dim3(kB, kT, 2), 256, 0, stream>>>(x, Wx_f, Wx_b, xst);

    lnlstm_kernel<<<kTotalWG, 256, 65536, stream>>>(
        x, Wh_f, Wh_b, Wx_f, Wx_b, b_f, b_b,
        gx_f, bx_f, gx_b, bx_b, gh_f, bh_f, gh_b, bh_b,
        gc_f, bc_f, gc_b, bc_b,
        bar, gst, cst, xst, hbuf);

    head_kernel<<<dim3(kB), dim3(kZ), 0, stream>>>(hbuf, W_mu, b_mu, W_sig, b_sig, eps, (float*)d_out);
}
